// cross_scale_attention_80874234184226
// MI455X (gfx1250) — matrix operand order repaired and checked by reading, not compiled, not run on hardware
//
#include <hip/hip_runtime.h>

typedef __bf16 bf16;
typedef __attribute__((ext_vector_type(16))) __bf16 v16bf;
typedef __attribute__((ext_vector_type(8)))  __bf16 v8bf;
typedef __attribute__((ext_vector_type(8)))  float  v8f;

#define WMMA_BF16(a, b, c) \
  __builtin_amdgcn_wmma_f32_16x16x32_bf16(false, (a), false, (b), (short)0, (c), false, false)

// ---------- helpers ----------

__device__ __forceinline__ bf16 f2bf(float f) {
  union { float f; unsigned u; } v; v.f = f;
  unsigned r = (v.u + 0x7FFFu + ((v.u >> 16) & 1u)) >> 16;   // RNE
  union { unsigned short s; bf16 b; } o; o.s = (unsigned short)r;
  return o.b;
}

// A fragment (16x32 bf16, ISA 7.12.2): lane row m = lane&15, half = lane>>4.
// Elements 0..7  -> k = half*16 + e            (contiguous run)
// Elements 8..15 -> k = half*16 + e   (contiguous run)
__device__ __forceinline__ v16bf ld_a(const bf16* p0, const bf16* p1) {
  v8bf lo = *(const v8bf*)p0;
  v8bf hi = *(const v8bf*)p1;
  return __builtin_shufflevector(lo, hi, 0, 1, 2, 3, 4, 5, 6, 7,
                                 8, 9, 10, 11, 12, 13, 14, 15);
}
// B fragment (32x16 bf16): lane col n = lane&15; element e -> k = half*16 + e
// => one contiguous 16-element run.
__device__ __forceinline__ v16bf ld_b(const bf16* p) {
  v8bf lo = *(const v8bf*)p;
  v8bf hi = *(const v8bf*)(p + 8);
  return __builtin_shufflevector(lo, hi, 0, 1, 2, 3, 4, 5, 6, 7,
                                 8, 9, 10, 11, 12, 13, 14, 15);
}
// C/D f32 16x16: VGPR i, lane -> m = i + 8*(lane>>4), n = lane&15

// ---------- zero fill (u32 granularity) ----------
__global__ void zero_u32(unsigned* __restrict__ p, long n) {
  long i = (long)blockIdx.x * 256 + threadIdx.x;
  if (i < n) p[i] = 0u;
}

// ---------- NCHW f32 -> halo-padded NHWC bf16 (pad=1 ring of zeros) ----------
__global__ void pack_input(const float* __restrict__ X, bf16* __restrict__ Xcl,
                           int H, int W) {
  const int Hp = H + 2, Wp = W + 2;
  long i = (long)blockIdx.x * 256 + threadIdx.x;
  long total = (long)8 * Hp * Wp * 64;
  if (i >= total) return;
  int c = (int)(i & 63);
  long r = i >> 6;
  int xx = (int)(r % Wp); r /= Wp;
  int yy = (int)(r % Hp);
  int n = (int)(r / Hp);
  float v = 0.f;
  if (yy >= 1 && yy <= H && xx >= 1 && xx <= W)
    v = X[(((size_t)n * 64 + c) * H + (yy - 1)) * W + (xx - 1)];
  Xcl[i] = f2bf(v);
}

// ---------- weight pack: f32 [64,64,3,3] -> bf16 [tap=9][co=64][cin=64] ----------
__global__ void pack_w(const float* __restrict__ W, bf16* __restrict__ Wt) {
  int i = blockIdx.x * 256 + threadIdx.x;
  if (i >= 9 * 64 * 64) return;
  int c = i & 63;
  int r = i >> 6;
  int co = r & 63, t = r >> 6;
  Wt[i] = f2bf(W[((co * 64 + c) * 3 + t / 3) * 3 + t % 3]);
}

// ---------- 3x3 conv + bias + LeakyReLU(0.01), tap-decomposed implicit GEMM ----
// Xcl: halo-padded NHWC bf16 [(H+2)(W+2)][64]; Wt: [9][64][64].
// Output either bf16 NHWC square buffer of pitch `opitch` (Ocl != null) or
// f32 NCHW (Of32). M=16-pixel tile per wave, N=64 (4 tiles), K=9 taps x 64.
__global__ __launch_bounds__(256)
void conv3_wmma(const bf16* __restrict__ Xcl, int H, int W,
                const bf16* __restrict__ Wt, const float* __restrict__ bias,
                bf16* __restrict__ Ocl, int opitch, float* __restrict__ Of32) {
  const int lane = threadIdx.x & 31;
  const int wave = threadIdx.x >> 5;
  const int half = lane >> 4, nl = lane & 15;
  const int n = blockIdx.y;
  const int Wp = W + 2;
  const int mtile = blockIdx.x * 8 + wave;
  const int pix = mtile * 16 + nl;
  const int py = pix / W, px = pix % W;
  const bf16* Xn = Xcl + ((size_t)n * (H + 2) * Wp + (size_t)py * Wp + px) * 64;
  v8f acc[4] = {};
#pragma unroll
  for (int t = 0; t < 9; ++t) {
    const bf16* ab = Xn + ((t / 3) * Wp + (t % 3)) * 64;
    if (t < 8)
      __builtin_prefetch(Xn + (((t + 1) / 3) * Wp + ((t + 1) % 3)) * 64, 0, 0);
#pragma unroll
    for (int c0 = 0; c0 < 64; c0 += 32) {
      v16bf a = ld_a(ab + c0 + half * 16, ab + c0 + half * 16 + 8);
#pragma unroll
      for (int nt = 0; nt < 4; ++nt) {
        v16bf b = ld_b(Wt + ((size_t)(t * 64 + nt * 16 + nl) * 64 + c0 + half * 16));
        acc[nt] = WMMA_BF16(a, b, acc[nt]);
      }
    }
  }
#pragma unroll
  for (int nt = 0; nt < 4; ++nt) {
    int co = nt * 16 + nl;
    float bb = bias[co];
#pragma unroll
    for (int i = 0; i < 8; ++i) {
      int m = mtile * 16 + i + 8 * half;
      float v = acc[nt][i] + bb;
      v = v > 0.f ? v : 0.01f * v;
      if (Ocl) {
        int my = m / W, mx = m % W;
        Ocl[(((size_t)n * opitch + my) * opitch + mx) * 64 + co] = f2bf(v);
      } else {
        Of32[((size_t)n * 64 + co) * (size_t)(H * W) + m] = v;
      }
    }
  }
}

// ---------- key patch sumsq per (n,l) over 25 taps x 64 ch ----------
__global__ void k_sumsq(const bf16* __restrict__ Kcl, float* __restrict__ Ssq) {
  __shared__ float red[256];
  int l = blockIdx.x, n = blockIdx.y, tid = threadIdx.x;
  int py = l >> 4, px = l & 15;
  float s = 0.f;
  for (int idx = tid; idx < 1600; idx += 256) {
    int c = idx & 63, t = idx >> 6;
    union { bf16 b; unsigned short u; } w;
    w.b = Kcl[((size_t)n * 4225 + (4 * py + t / 5) * 65 + 4 * px + t % 5) * 64 + c];
    union { unsigned u; float f; } cv; cv.u = (unsigned)w.u << 16;
    s += cv.f * cv.f;
  }
  red[tid] = s; __syncthreads();
  for (int st = 128; st > 0; st >>= 1) {
    if (tid < st) red[tid] += red[tid + st];
    __syncthreads();
  }
  if (tid == 0) Ssq[n * 256 + l] = red[0];
}

__global__ void k_max(const float* __restrict__ Ssq, float* __restrict__ Kmax) {
  __shared__ float red[256];
  int n = blockIdx.x, tid = threadIdx.x;
  red[tid] = sqrtf(Ssq[n * 256 + tid]);
  __syncthreads();
  for (int st = 128; st > 0; st >>= 1) {
    if (tid < st) red[tid] = fmaxf(red[tid], red[tid + st]);
    __syncthreads();
  }
  if (tid == 0) Kmax[n] = red[0];
}

// ---------- normalized key pack: bf16 [n][tap=25][l=256][c=64] ----------
__global__ void pack_kn(const bf16* __restrict__ Kcl, const float* __restrict__ Kmax,
                        bf16* __restrict__ KnT) {
  int i = blockIdx.x * 256 + threadIdx.x;
  if (i >= 8 * 25 * 256 * 64) return;
  int c = i & 63;
  int r = i >> 6;
  int l = r & 255; r >>= 8;
  int t = r % 25, n = r / 25;
  union { bf16 b; unsigned short u; } w;
  w.b = Kcl[((size_t)n * 4225 + (4 * (l >> 4) + t / 5) * 65 + 4 * (l & 15) + t % 5) * 64 + c];
  union { unsigned u; float f; } cv; cv.u = (unsigned)w.u << 16;
  KnT[i] = f2bf(cv.f / Kmax[n]);
}

// ---------- value patch pack: bf16 [n][tap=25][b=64][l=256] ----------
__global__ void pack_vp(const bf16* __restrict__ Vcl, bf16* __restrict__ VpT) {
  int i = blockIdx.x * 256 + threadIdx.x;
  if (i >= 8 * 25 * 64 * 256) return;
  int l = i & 255;
  int r = i >> 8;
  int b = r & 63; r >>= 6;
  int t = r % 25, n = r / 25;
  VpT[i] = Vcl[((size_t)n * 4225 + (4 * (l >> 4) + t / 5) * 65 + 4 * (l & 15) + t % 5) * 64 + b];
}

// ---------- attention scores: M=4096 positions, N=L=256, K=25 taps x 64 ch ----
// Qcl: bf16 [n][257][257][64] (row/col 256 zero = "same" pad); KnT as above.
// Wsc: f32 [n][pix=4096][l=256]. 8 waves share the M tile; each owns 2 L-tiles.
__global__ __launch_bounds__(256)
void attn_scores_wmma(const bf16* __restrict__ Qcl, const bf16* __restrict__ KnT,
                      float* __restrict__ Wsc) {
  const int lane = threadIdx.x & 31;
  const int wave = threadIdx.x >> 5;
  const int half = lane >> 4, nl = lane & 15;
  const int n = blockIdx.y;
  const int mtile = blockIdx.x;
  const int pix = mtile * 16 + nl;
  const int wy = pix >> 6, wx = pix & 63;
  const bf16* Qn = Qcl + ((size_t)n * 257 * 257 + (size_t)(4 * wy) * 257 + 4 * wx) * 64;
  const bf16* Kn0 = KnT + (size_t)n * 25 * 256 * 64;
  v8f acc[2] = {};
#pragma unroll
  for (int t = 0; t < 25; ++t) {
    const bf16* ab = Qn + ((t / 5) * 257 + (t % 5)) * 64;
#pragma unroll
    for (int c0 = 0; c0 < 64; c0 += 32) {
      v16bf a = ld_a(ab + c0 + half * 16, ab + c0 + half * 16 + 8);
#pragma unroll
      for (int s = 0; s < 2; ++s) {
        int l = (wave * 2 + s) * 16 + nl;
        v16bf b = ld_b(Kn0 + ((size_t)(t * 256 + l)) * 64 + c0 + half * 16);
        acc[s] = WMMA_BF16(a, b, acc[s]);
      }
    }
  }
#pragma unroll
  for (int s = 0; s < 2; ++s) {
    int l = (wave * 2 + s) * 16 + nl;
#pragma unroll
    for (int i = 0; i < 8; ++i) {
      int m = mtile * 16 + i + 8 * half;
      Wsc[((size_t)n * 4096 + m) * 256 + l] = acc[s][i];
    }
  }
}

// ---------- softmax over L (x10) -> bf16 [n][pofs(65x65)][l] (halo stays 0) ----
__global__ __launch_bounds__(256)
void softmax_pack(const float* __restrict__ Wsc, bf16* __restrict__ WnT) {
  int p = blockIdx.x * 256 + threadIdx.x;    // 0..4095
  int n = blockIdx.y;
  const float* w = Wsc + ((size_t)n * 4096 + p) * 256;
  int pofs = (p >> 6) * 65 + (p & 63);
  bf16* o = WnT + ((size_t)n * 4225 + pofs) * 256;
  float mx = -3.4e38f;
  for (int l = 0; l < 256; ++l) mx = fmaxf(mx, w[l]);
  float s = 0.f;
  for (int l = 0; l < 256; ++l) s += __expf(10.f * (w[l] - mx));
  float inv = 1.f / s;
  for (int l = 0; l < 256; ++l) o[l] = f2bf(__expf(10.f * (w[l] - mx)) * inv);
}

// ---------- transposed conv (stride4/ks5/pad1/outpad1) via residue-class GEMMs
// out[b,Y,X] = (1/6) sum_l sum_taps wn[l,y,x] * vp[l,b,4-ky,4-kx],
// ky ≡ (3-Y%4) mod 4 (+4 when 0), y=(Y+ky-3)/4.  M tiles grouped by residue so
// taps are tile-uniform; wnT's zero halo (row/col 64) absorbs boundary taps.
// Result written as halo-padded NHWC bf16 [258][258][64] for the final conv.
__global__ __launch_bounds__(256)
void attn_apply_wmma(const bf16* __restrict__ WnT, const bf16* __restrict__ VpT,
                     bf16* __restrict__ Rcl) {
  const int lane = threadIdx.x & 31;
  const int wave = threadIdx.x >> 5;
  const int half = lane >> 4, nl = lane & 15;
  const int n = blockIdx.y;
  const int mt = blockIdx.x * 8 + wave;      // 0..4095
  const int res_cls = mt >> 8;
  const int twr = mt & 255;
  const int ry = res_cls >> 2, rx = res_cls & 3;
  const int pi = twr * 16 + nl;
  const int yc = pi >> 6, xc = pi & 63;
  const int ky0 = (3 - ry) & 3, kx0 = (3 - rx) & 3;
  const int nty = (ky0 == 0) ? 2 : 1, ntx = (kx0 == 0) ? 2 : 1;
  const bf16* wn = WnT + (size_t)n * 4225 * 256;
  const bf16* vp = VpT + (size_t)n * 25 * 64 * 256;
  v8f acc[4] = {};
  for (int ty = 0; ty < nty; ++ty) {
    int ky = ky0 + 4 * ty;
    for (int tx = 0; tx < ntx; ++tx) {
      int kx = kx0 + 4 * tx;
      const bf16* ab = wn + ((size_t)(yc + ty) * 65 + (xc + tx)) * 256;
      const bf16* vb = vp + (size_t)((4 - ky) * 5 + (4 - kx)) * 64 * 256;
      __builtin_prefetch(ab, 0, 0);
#pragma unroll
      for (int k0 = 0; k0 < 256; k0 += 32) {
        v16bf a = ld_a(ab + k0 + half * 16, ab + k0 + half * 16 + 8);
#pragma unroll
        for (int nt = 0; nt < 4; ++nt) {
          v16bf b = ld_b(vb + (size_t)(nt * 16 + nl) * 256 + k0 + half * 16);
          acc[nt] = WMMA_BF16(a, b, acc[nt]);
        }
      }
    }
  }
#pragma unroll
  for (int nt = 0; nt < 4; ++nt) {
    int co = nt * 16 + nl;
#pragma unroll
    for (int i = 0; i < 8; ++i) {
      int m = twr * 16 + i + 8 * half;
      int Y = 4 * (m >> 6) + ry, X = 4 * (m & 63) + rx;
      Rcl[(((size_t)n * 258 + (Y + 1)) * 258 + (X + 1)) * 64 + co] =
          f2bf(acc[nt][i] * (1.f / 6.f));
    }
  }
}

// ---------- driver ----------
extern "C" void kernel_launch(void* const* d_in, const int* in_sizes, int n_in,
                              void* d_out, int out_size, void* d_ws, size_t ws_size,
                              hipStream_t stream) {
  const float* ms   = (const float*)d_in[0];
  const float* pan  = (const float*)d_in[1];
  const float* pan2 = (const float*)d_in[2];
  const float* Wq   = (const float*)d_in[3];
  const float* bq   = (const float*)d_in[4];
  const float* Wk   = (const float*)d_in[5];
  const float* bk   = (const float*)d_in[6];
  const float* Wv   = (const float*)d_in[7];
  const float* bv   = (const float*)d_in[8];
  const float* Wr   = (const float*)d_in[9];
  const float* br   = (const float*)d_in[10];
  float* out = (float*)d_out;
  (void)in_sizes; (void)n_in; (void)out_size; (void)ws_size;

  char* ws = (char*)d_ws;
  size_t off = 0;
  auto alloc = [&](size_t bytes) -> void* {
    void* p = ws + off;
    off += (bytes + 255) & ~(size_t)255;
    return p;
  };
  bf16* pancl  = (bf16*)alloc((size_t)8 * 258 * 258 * 64 * 2);
  bf16* pan2cl = (bf16*)alloc((size_t)8 * 66 * 66 * 64 * 2);
  bf16* mscl   = (bf16*)alloc((size_t)8 * 66 * 66 * 64 * 2);
  bf16* qcl    = (bf16*)alloc((size_t)8 * 257 * 257 * 64 * 2);
  bf16* kcl    = (bf16*)alloc((size_t)8 * 65 * 65 * 64 * 2);
  bf16* vcl    = (bf16*)alloc((size_t)8 * 65 * 65 * 64 * 2);
  bf16* rescl  = (bf16*)alloc((size_t)8 * 258 * 258 * 64 * 2);
  float* wsc   = (float*)alloc((size_t)8 * 4096 * 256 * 4);
  bf16* wnT    = (bf16*)alloc((size_t)8 * 4225 * 256 * 2);
  bf16* knT    = (bf16*)alloc((size_t)8 * 25 * 256 * 64 * 2);
  bf16* vpT    = (bf16*)alloc((size_t)8 * 25 * 64 * 256 * 2);
  bf16* wqt    = (bf16*)alloc(9 * 64 * 64 * 2);
  bf16* wkt    = (bf16*)alloc(9 * 64 * 64 * 2);
  bf16* wvt    = (bf16*)alloc(9 * 64 * 64 * 2);
  bf16* wrt    = (bf16*)alloc(9 * 64 * 64 * 2);
  float* ssq   = (float*)alloc(8 * 256 * 4);
  float* kmx   = (float*)alloc(8 * 4);

  auto zero = [&](void* p, long bytes) {
    long n = bytes / 4;
    zero_u32<<<(unsigned)((n + 255) / 256), 256, 0, stream>>>((unsigned*)p, n);
  };
  // zero padded buffers (halos survive; interiors are overwritten below)
  zero(qcl,   (long)8 * 257 * 257 * 64 * 2);
  zero(kcl,   (long)8 * 65 * 65 * 64 * 2);
  zero(vcl,   (long)8 * 65 * 65 * 64 * 2);
  zero(wnT,   (long)8 * 4225 * 256 * 2);
  zero(rescl, (long)8 * 258 * 258 * 64 * 2);

  pack_w<<<(9 * 64 * 64 + 255) / 256, 256, 0, stream>>>(Wq, wqt);
  pack_w<<<(9 * 64 * 64 + 255) / 256, 256, 0, stream>>>(Wk, wkt);
  pack_w<<<(9 * 64 * 64 + 255) / 256, 256, 0, stream>>>(Wv, wvt);
  pack_w<<<(9 * 64 * 64 + 255) / 256, 256, 0, stream>>>(Wr, wrt);

  {
    long t1 = (long)8 * 258 * 258 * 64;
    pack_input<<<(unsigned)((t1 + 255) / 256), 256, 0, stream>>>(pan, pancl, 256, 256);
    long t2 = (long)8 * 66 * 66 * 64;
    pack_input<<<(unsigned)((t2 + 255) / 256), 256, 0, stream>>>(pan2, pan2cl, 64, 64);
    pack_input<<<(unsigned)((t2 + 255) / 256), 256, 0, stream>>>(ms, mscl, 64, 64);
  }

  conv3_wmma<<<dim3(32, 8), 256, 0, stream>>>(pan2cl, 64, 64, wkt, bk, kcl, 65, nullptr);
  conv3_wmma<<<dim3(32, 8), 256, 0, stream>>>(mscl,   64, 64, wvt, bv, vcl, 65, nullptr);
  conv3_wmma<<<dim3(512, 8), 256, 0, stream>>>(pancl, 256, 256, wqt, bq, qcl, 257, nullptr);

  k_sumsq<<<dim3(256, 8), 256, 0, stream>>>(kcl, ssq);
  k_max<<<8, 256, 0, stream>>>(ssq, kmx);
  pack_kn<<<(8 * 25 * 256 * 64 + 255) / 256, 256, 0, stream>>>(kcl, kmx, knT);
  pack_vp<<<(8 * 25 * 64 * 256 + 255) / 256, 256, 0, stream>>>(vcl, vpT);

  attn_scores_wmma<<<dim3(256, 8), 256, 0, stream>>>(qcl, knT, wsc);
  softmax_pack<<<dim3(16, 8), 256, 0, stream>>>(wsc, wnT);
  attn_apply_wmma<<<dim3(512, 8), 256, 0, stream>>>(wnT, vpT, rescl);

  conv3_wmma<<<dim3(512, 8), 256, 0, stream>>>(rescl, 256, 256, wrt, br, nullptr, 0, out);
}
